// RATLayer_16372415332769
// MI455X (gfx1250) — matrix operand order repaired and checked by reading, not compiled, not run on hardware
//
#include <hip/hip_runtime.h>

// RAT layer for MI455X (gfx1250).
// - bf16 WMMA (v_wmma_f32_16x16x32_bf16) for every GEMM-shaped op, fp32 accum
// - bf16 storage for the dominant HBM streams (rk/rv/q/k/v/probs)
// - GEMM: double-buffered GLOBAL_LOAD_ASYNC_TO_LDS_B128 pipeline (ASYNCcnt)
// - weights pre-transposed to bf16 Wt[N,K] once so B-staging is a pure DMA
// B=4 L=512 DM=1024 H=16 DK=64 DR=64 DF=4096

#define DEV __device__ __forceinline__

typedef __attribute__((ext_vector_type(16))) __bf16 v16bf;
typedef __attribute__((ext_vector_type(8)))  __bf16 v8bf;
typedef __attribute__((ext_vector_type(8)))  float  v8f;

DEV v16bf cat8(v8bf lo, v8bf hi) {
  return __builtin_shufflevector(lo, hi, 0,1,2,3,4,5,6,7,8,9,10,11,12,13,14,15);
}
DEV v8bf cvt8f(const float* p) {
  v8bf r;
#pragma unroll
  for (int i = 0; i < 8; i++) r[i] = (__bf16)p[i];
  return r;
}
DEV v16bf cvt16f(const float* p0, const float* p1) {
  v16bf r;
#pragma unroll
  for (int i = 0; i < 8; i++) { r[i] = (__bf16)p0[i]; r[8 + i] = (__bf16)p1[i]; }
  return r;
}
DEV v8f wmma_bf16(v16bf a, v16bf b, v8f c) {
  return __builtin_amdgcn_wmma_f32_16x16x32_bf16(false, a, false, b, (short)0, c, false, false);
}
// Async DMA: 16 bytes global -> LDS, tracked by ASYNCcnt.
DEV void async_copy16(unsigned lds_off, const void* gaddr) {
  asm volatile("global_load_async_to_lds_b128 %0, %1, off"
               :: "v"(lds_off), "v"(gaddr) : "memory");
}
#define WAIT_ASYNC(n) asm volatile("s_wait_asynccnt " #n ::: "memory")

// ---------------------------------------------------------------------------
// One-time weight prep: W[K,N] fp32 -> Wt[N,K] bf16 (32x32 tiles via LDS).
// ---------------------------------------------------------------------------
__global__ __launch_bounds__(256)
void wprep_kernel(const float* __restrict__ W, __bf16* __restrict__ Wt, int K, int N) {
  __shared__ float t[32][33];
  const int k0 = blockIdx.x * 32, n0 = blockIdx.y * 32;
  const int tx = threadIdx.x & 31, ty = threadIdx.x >> 5;   // ty: 0..7
#pragma unroll
  for (int i = 0; i < 4; i++)
    t[ty + 8 * i][tx] = W[(size_t)(k0 + ty + 8 * i) * N + n0 + tx];
  __syncthreads();
#pragma unroll
  for (int i = 0; i < 4; i++)
    Wt[(size_t)(n0 + ty + 8 * i) * K + k0 + tx] = (__bf16)t[tx][ty + 8 * i];
}

// ---------------------------------------------------------------------------
// LayerNorm over rows of 1024. One block (256 thr) per row.
// ---------------------------------------------------------------------------
__global__ __launch_bounds__(256)
void ln_kernel(const float* __restrict__ x, const float* __restrict__ g,
               const float* __restrict__ b, float* __restrict__ y) {
  const int row = blockIdx.x;
  const float* xr = x + (size_t)row * 1024;
  float* yr = y + (size_t)row * 1024;
  float v[4], s = 0.f, ss = 0.f;
#pragma unroll
  for (int i = 0; i < 4; i++) {
    float t = xr[threadIdx.x + 256 * i];
    v[i] = t; s += t; ss += t * t;
  }
#pragma unroll
  for (int m = 16; m > 0; m >>= 1) { s += __shfl_xor(s, m, 32); ss += __shfl_xor(ss, m, 32); }
  __shared__ float red[2][8];
  const int wid = threadIdx.x >> 5, lane = threadIdx.x & 31;
  if (lane == 0) { red[0][wid] = s; red[1][wid] = ss; }
  __syncthreads();
  s = 0.f; ss = 0.f;
#pragma unroll
  for (int i = 0; i < 8; i++) { s += red[0][i]; ss += red[1][i]; }
  const float mean = s * (1.f / 1024.f);
  const float var  = ss * (1.f / 1024.f) - mean * mean;
  const float inv  = rsqrtf(var + 1e-6f);
#pragma unroll
  for (int i = 0; i < 4; i++) {
    int c = threadIdx.x + 256 * i;
    yr[c] = (v[i] - mean) * inv * g[c] + b[c];
  }
}

// ---------------------------------------------------------------------------
// GEMM: out = act(A[M,K]*W[K,N] + bias (+res)).  A fp32, Wt = bf16 [N,K].
// Block 64x128, 8 waves (2x4), 32x32/wave. Double-buffered async LDS pipeline.
// ---------------------------------------------------------------------------
template <bool RES, bool RELU, bool OUTBF>
__global__ __launch_bounds__(256)
void gemm_kernel(const float* __restrict__ A, const __bf16* __restrict__ Wt,
                 const float* __restrict__ bias, const float* __restrict__ res,
                 float* __restrict__ outF, __bf16* __restrict__ outB,
                 int M, int N, int K) {
  __shared__ __align__(16) float  ldsAf[2][64][32];   // fp32 A tiles (DMA'd)
  __shared__ __align__(16) __bf16 ldsB[2][128][32];   // bf16 W tiles [n][k] (DMA'd)
  const int tid  = threadIdx.x;
  const int wid  = tid >> 5, lane = tid & 31;
  const int wm   = wid >> 2, wn = wid & 3;
  const int m0   = blockIdx.y * 64, n0 = blockIdx.x * 128;
  const int kbo  = (lane >> 4) * 8;
  const int kb2  = (lane >> 4) * 16;
  v8f acc[2][2] = {};

  const int sr = tid >> 2, sc = (tid & 3) * 8;        // A stage map (8 floats)
  const int sn = tid >> 1, sk = (tid & 1) * 16;       // B stage map (16 bf16)
  auto stage = [&](int bb, int kk) {
    const float* ga = A + (size_t)(m0 + sr) * K + kk + sc;
    async_copy16((unsigned)(size_t)&ldsAf[bb][sr][sc],     ga);
    async_copy16((unsigned)(size_t)&ldsAf[bb][sr][sc + 4], ga + 4);
    const __bf16* gb = Wt + (size_t)(n0 + sn) * K + kk + sk;
    async_copy16((unsigned)(size_t)&ldsB[bb][sn][sk],     gb);
    async_copy16((unsigned)(size_t)&ldsB[bb][sn][sk + 8], gb + 8);
  };

  stage(0, 0);
  int it = 0;
  for (int k0 = 0; k0 < K; k0 += 32, it ^= 1) {
    const bool more = (k0 + 32 < K);
    if (more) { stage(it ^ 1, k0 + 32); WAIT_ASYNC(4); }
    else      { WAIT_ASYNC(0); }
    __syncthreads();
    v16bf afr[2], bfr[2];
    const int mrow = wm * 32 + (lane & 15);
    const int bcol = wn * 32 + (lane & 15);
#pragma unroll
    for (int mt = 0; mt < 2; mt++)
      afr[mt] = cvt16f(&ldsAf[it][mrow + mt * 16][kb2],
                       &ldsAf[it][mrow + mt * 16][kb2 + 8]);
#pragma unroll
    for (int nt = 0; nt < 2; nt++)
      bfr[nt] = cat8(*(const v8bf*)&ldsB[it][bcol + nt * 16][kb2],
                     *(const v8bf*)&ldsB[it][bcol + nt * 16][kb2 + 8]);
#pragma unroll
    for (int mt = 0; mt < 2; mt++)
#pragma unroll
      for (int nt = 0; nt < 2; nt++)
        acc[mt][nt] = wmma_bf16(afr[mt], bfr[nt], acc[mt][nt]);
    __syncthreads();
  }
#pragma unroll
  for (int mt = 0; mt < 2; mt++)
#pragma unroll
    for (int nt = 0; nt < 2; nt++) {
      const int col = n0 + wn * 32 + nt * 16 + (lane & 15);
      const float bv = bias[col];
#pragma unroll
      for (int j = 0; j < 8; j++) {
        const int row = m0 + wm * 32 + mt * 16 + (lane >> 4) * 8 + j;
        float val = acc[mt][nt][j] + bv;
        if (RES)  val += res[(size_t)row * N + col];
        if (RELU) val = fmaxf(val, 0.f);
        if (OUTBF) outB[(size_t)row * N + col] = (__bf16)val;
        else       outF[(size_t)row * N + col] = val;
      }
    }
}

// ---------------------------------------------------------------------------
// rk/rv = R[M,64] @ Wr[64,64] + br -> bf16. Streaming, HBM-bound.
// ---------------------------------------------------------------------------
__global__ __launch_bounds__(256)
void rproj_kernel(const float* __restrict__ R, const float* __restrict__ Wr,
                  const float* __restrict__ br, __bf16* __restrict__ out) {
  __shared__ __align__(16) __bf16 ldsW[64][64];       // transposed [n][k]
  __shared__ float ldsb[64];
  const int tid = threadIdx.x;
  { // vectorized staging: each thread builds 16 contiguous k's of one n-row
    const int n = tid & 63, kk0 = (tid >> 6) * 16;
    v8bf w0, w1;
#pragma unroll
    for (int i = 0; i < 8; i++) {
      w0[i] = (__bf16)Wr[(size_t)(kk0 + i) * 64 + n];
      w1[i] = (__bf16)Wr[(size_t)(kk0 + 8 + i) * 64 + n];
    }
    *(v8bf*)&ldsW[n][kk0] = w0;
    *(v8bf*)&ldsW[n][kk0 + 8] = w1;
  }
  if (tid < 64) ldsb[tid] = br[tid];
  __syncthreads();
  const int wid = tid >> 5, lane = tid & 31;
  const int kbo = (lane >> 4) * 8, kb2 = (lane >> 4) * 16;
  const size_t mbase = ((size_t)blockIdx.x * 8 + wid) * 16;
  const float* arow = R + (mbase + (lane & 15)) * 64;
  __builtin_prefetch(arow + 128 * 64, 0, 1);
  v8f acc[4] = {};
#pragma unroll
  for (int ks = 0; ks < 2; ks++) {
    const int k0 = ks * 32;
    v16bf af = cat8(cvt8f(arow + k0 + kb2), cvt8f(arow + k0 + kb2 + 8));
#pragma unroll
    for (int nt = 0; nt < 4; nt++) {
      const int bn = nt * 16 + (lane & 15);
      v16bf bf_ = cat8(*(const v8bf*)&ldsW[bn][k0 + kb2],
                       *(const v8bf*)&ldsW[bn][k0 + kb2 + 8]);
      acc[nt] = wmma_bf16(af, bf_, acc[nt]);
    }
  }
#pragma unroll
  for (int nt = 0; nt < 4; nt++) {
    const int col = nt * 16 + (lane & 15);
    const float bb = ldsb[col];
#pragma unroll
    for (int j = 0; j < 8; j++)
      out[(mbase + (lane >> 4) * 8 + j) * 64 + col] = (__bf16)(acc[nt][j] + bb);
  }
}

// ---------------------------------------------------------------------------
// scores[b,h,q,k] = q . k  (per (b,h); block = 16 q x 512 k)
// ---------------------------------------------------------------------------
__global__ __launch_bounds__(256)
void scores_qk_kernel(const __bf16* __restrict__ qb, const __bf16* __restrict__ kmat,
                      float* __restrict__ scores) {
  const int bh = blockIdx.x, b = bh >> 4, h = bh & 15;
  const int q0 = blockIdx.y * 16;
  const int tid = threadIdx.x, wid = tid >> 5, lane = tid & 31;
  const int n0 = wid * 64;
  const int kbo = (lane >> 4) * 8, kb2 = (lane >> 4) * 16;
  const __bf16* qbase = qb + ((size_t)b * 512 + q0) * 1024 + h * 64;
  const __bf16* kbase = kmat + (size_t)b * 512 * 1024 + h * 64;
  v8f acc[4] = {};
#pragma unroll
  for (int ks = 0; ks < 2; ks++) {
    const int d0 = ks * 32;
    const __bf16* ap = qbase + (size_t)(lane & 15) * 1024 + d0 + kb2;
    v16bf af = cat8(*(const v8bf*)ap, *(const v8bf*)(ap + 8));
#pragma unroll
    for (int nt = 0; nt < 4; nt++) {
      const int col = n0 + nt * 16 + (lane & 15);
      const __bf16* bp = kbase + (size_t)col * 1024 + d0 + kb2;
      v16bf bf_ = cat8(*(const v8bf*)bp, *(const v8bf*)(bp + 8));
      acc[nt] = wmma_bf16(af, bf_, acc[nt]);
    }
  }
  float* srow = scores + ((size_t)bh * 512 + q0) * 512;
#pragma unroll
  for (int nt = 0; nt < 4; nt++) {
    const int col = n0 + nt * 16 + (lane & 15);
#pragma unroll
    for (int j = 0; j < 8; j++)
      srow[(size_t)((lane >> 4) * 8 + j) * 512 + col] = acc[nt][j];
  }
}

// ---------------------------------------------------------------------------
// scores[b,:,q,:] += Q(16h x 64d) @ rk[b,q]^T. One block per (b,q).
// ---------------------------------------------------------------------------
__global__ __launch_bounds__(256)
void scores_qr_kernel(const __bf16* __restrict__ qb, const __bf16* __restrict__ rk,
                      float* __restrict__ scores) {
  const int bq = blockIdx.x, b = bq >> 9, q = bq & 511;
  const int tid = threadIdx.x, wid = tid >> 5, lane = tid & 31;
  const int n0 = wid * 64;
  const int kbo = (lane >> 4) * 8, kb2 = (lane >> 4) * 16;
  const __bf16* qrow  = qb + ((size_t)b * 512 + q) * 1024;
  const __bf16* rbase = rk + (size_t)bq * 512 * 64;
  v8f acc[4] = {};
#pragma unroll
  for (int ks = 0; ks < 2; ks++) {
    const int d0 = ks * 32;
    const __bf16* ap = qrow + (size_t)(lane & 15) * 64 + d0 + kb2;
    v16bf af = cat8(*(const v8bf*)ap, *(const v8bf*)(ap + 8));
#pragma unroll
    for (int nt = 0; nt < 4; nt++) {
      const int col = n0 + nt * 16 + (lane & 15);
      const __bf16* bp = rbase + (size_t)col * 64 + d0 + kb2;
      v16bf bf_ = cat8(*(const v8bf*)bp, *(const v8bf*)(bp + 8));
      acc[nt] = wmma_bf16(af, bf_, acc[nt]);
    }
  }
#pragma unroll
  for (int nt = 0; nt < 4; nt++) {
    const int col = n0 + nt * 16 + (lane & 15);
#pragma unroll
    for (int j = 0; j < 8; j++) {
      const int h = (lane >> 4) * 8 + j;
      scores[(((size_t)b * 16 + h) * 512 + q) * 512 + col] += acc[nt][j];
    }
  }
}

// ---------------------------------------------------------------------------
// softmax (scale 1/8, mask), fp32 -> bf16 probs. One wave per row.
// ---------------------------------------------------------------------------
__global__ __launch_bounds__(256)
void softmax_kernel(const float* __restrict__ scores, const unsigned char* __restrict__ mask,
                    __bf16* __restrict__ probs) {
  const int row  = blockIdx.x * 8 + (threadIdx.x >> 5);
  const int lane = threadIdx.x & 31;
  const int b = row >> 13, q = row & 511;
  const unsigned char* mrow = mask + ((size_t)b * 512 + q) * 512;
  const float* srow = scores + (size_t)row * 512;
  __bf16* prow = probs + (size_t)row * 512;
  float vals[16], mx = -1e30f;
#pragma unroll
  for (int i = 0; i < 16; i++) {
    const int c = lane + 32 * i;
    float s = srow[c] * 0.125f;
    if (mrow[c]) s = -1e9f;
    vals[i] = s; mx = fmaxf(mx, s);
  }
#pragma unroll
  for (int m = 16; m > 0; m >>= 1) mx = fmaxf(mx, __shfl_xor(mx, m, 32));
  float sum = 0.f;
#pragma unroll
  for (int i = 0; i < 16; i++) { float e = __expf(vals[i] - mx); vals[i] = e; sum += e; }
#pragma unroll
  for (int m = 16; m > 0; m >>= 1) sum += __shfl_xor(sum, m, 32);
  const float inv = 1.f / sum;
#pragma unroll
  for (int i = 0; i < 16; i++) prow[lane + 32 * i] = (__bf16)(vals[i] * inv);
}

// ---------------------------------------------------------------------------
// ctx = a @ v per (b,h). Block: 128 q x 64 d; V tile transposed in LDS.
// ---------------------------------------------------------------------------
__global__ __launch_bounds__(256)
void ctx_av_kernel(const __bf16* __restrict__ probs, const __bf16* __restrict__ vb,
                   float* __restrict__ ctx) {
  __shared__ __align__(16) __bf16 ldsV[64][32];       // transposed [d][k]
  const int bh = blockIdx.x, b = bh >> 4, h = bh & 15;
  const int q0 = blockIdx.y * 128;
  const int tid = threadIdx.x, wid = tid >> 5, lane = tid & 31;
  const int kbo = (lane >> 4) * 8, kb2 = (lane >> 4) * 16;
  const __bf16* prow0 = probs + ((size_t)bh * 512 + q0 + wid * 16) * 512;
  const __bf16* vbase = vb + (size_t)b * 512 * 1024 + h * 64;
  const int sd = tid & 63, sk = (tid >> 6) * 8;       // stage map: 8 k's of one d-row
  v8f acc[4] = {};
  for (int k0 = 0; k0 < 512; k0 += 32) {
    { // vectorized transpose staging: one 16B LDS store per thread
      v8bf vv;
#pragma unroll
      for (int i = 0; i < 8; i++) vv[i] = vbase[(size_t)(k0 + sk + i) * 1024 + sd];
      *(v8bf*)&ldsV[sd][sk] = vv;
    }
    __syncthreads();
    const __bf16* ap = prow0 + (size_t)(lane & 15) * 512 + k0 + kb2;
    v16bf af = cat8(*(const v8bf*)ap, *(const v8bf*)(ap + 8));
#pragma unroll
    for (int nt = 0; nt < 4; nt++) {
      const int bn = nt * 16 + (lane & 15);
      v16bf bf_ = cat8(*(const v8bf*)&ldsV[bn][kb2], *(const v8bf*)&ldsV[bn][kb2 + 8]);
      acc[nt] = wmma_bf16(af, bf_, acc[nt]);
    }
    __syncthreads();
  }
#pragma unroll
  for (int nt = 0; nt < 4; nt++) {
    const int d = nt * 16 + (lane & 15);
#pragma unroll
    for (int j = 0; j < 8; j++) {
      const int qq = q0 + wid * 16 + (lane >> 4) * 8 + j;
      ctx[((size_t)b * 512 + qq) * 1024 + h * 64 + d] = acc[nt][j];
    }
  }
}

// ---------------------------------------------------------------------------
// ctx[b,q,:,:] += a2(16h x 512k) @ rv[b,q](512k x 64d). One wave per (b,q).
// ---------------------------------------------------------------------------
__global__ __launch_bounds__(256)
void ctx_ar_kernel(const __bf16* __restrict__ probs, const __bf16* __restrict__ rv,
                   float* __restrict__ ctx) {
  const int tid = threadIdx.x, wid = tid >> 5, lane = tid & 31;
  const int bq = blockIdx.x * 8 + wid, b = bq >> 9, q = bq & 511;
  const int kbo = (lane >> 4) * 8, kb2 = (lane >> 4) * 16;
  const __bf16* arow  = probs + (size_t)(b * 16 + (lane & 15)) * 262144 + (size_t)q * 512;
  const __bf16* rbase = rv + (size_t)bq * 512 * 64;
  v8f acc[4] = {};
  for (int k0 = 0; k0 < 512; k0 += 32) {
    const __bf16* ap = arow + k0 + kb2;
    v16bf af = cat8(*(const v8bf*)ap, *(const v8bf*)(ap + 8));
#pragma unroll
    for (int nt = 0; nt < 4; nt++) {
      const int dcol = nt * 16 + (lane & 15);
      v16bf bf_;
#pragma unroll
      for (int e = 0; e < 16; e++)
        bf_[e] = rbase[(size_t)(k0 + kb2 + e) * 64 + dcol];
      acc[nt] = wmma_bf16(af, bf_, acc[nt]);
    }
  }
#pragma unroll
  for (int nt = 0; nt < 4; nt++) {
    const int d = nt * 16 + (lane & 15);
#pragma unroll
    for (int j = 0; j < 8; j++) {
      const int h = (lane >> 4) * 8 + j;
      ctx[((size_t)b * 512 + q) * 1024 + h * 64 + d] += acc[nt][j];
    }
  }
}

// ---------------------------------------------------------------------------
extern "C" void kernel_launch(void* const* d_in, const int* in_sizes, int n_in,
                              void* d_out, int out_size, void* d_ws, size_t ws_size,
                              hipStream_t stream) {
  (void)in_sizes; (void)n_in; (void)out_size; (void)ws_size;
  const float* x   = (const float*)d_in[0];
  const float* r_k = (const float*)d_in[1];
  const float* r_v = (const float*)d_in[2];
  const unsigned char* mask = (const unsigned char*)d_in[3];
  const float* g1 = (const float*)d_in[4];  const float* b1 = (const float*)d_in[5];
  const float* Wq = (const float*)d_in[6];  const float* bq = (const float*)d_in[7];
  const float* Wk = (const float*)d_in[8];  const float* bk = (const float*)d_in[9];
  const float* Wv = (const float*)d_in[10]; const float* bv = (const float*)d_in[11];
  const float* Wr = (const float*)d_in[12]; const float* br = (const float*)d_in[13];
  const float* Wo = (const float*)d_in[14]; const float* bo = (const float*)d_in[15];
  const float* g2 = (const float*)d_in[16]; const float* b2 = (const float*)d_in[17];
  const float* W1 = (const float*)d_in[18]; const float* c1 = (const float*)d_in[19];
  const float* W2 = (const float*)d_in[20]; const float* c2 = (const float*)d_in[21];

  char* ws = (char*)d_ws;
  const size_t MiB = (size_t)1 << 20;
  float*  h      = (float*)(ws + 0);            // 8 MiB (reused as h2)
  __bf16* qb     = (__bf16*)(ws + 8   * MiB);   // 4 MiB
  __bf16* kb_    = (__bf16*)(ws + 12  * MiB);   // 4 MiB
  __bf16* vb     = (__bf16*)(ws + 16  * MiB);   // 4 MiB
  float*  ctx    = (float*)(ws + 20  * MiB);    // 8 MiB
  float*  outbuf = (float*)(ws + 28  * MiB);    // 8 MiB
  float*  scores = (float*)(ws + 36  * MiB);    // 64 MiB (reused as ffn1)
  __bf16* rkbf   = (__bf16*)(ws + 100 * MiB);   // 128 MiB (reused as probs)
  __bf16* rvbf   = (__bf16*)(ws + 228 * MiB);   // 128 MiB
  __bf16* wtq    = (__bf16*)(ws + 356 * MiB);   // 2 MiB each: wtq/wtk/wtv/wto
  __bf16* wtk    = (__bf16*)(ws + 358 * MiB);
  __bf16* wtv    = (__bf16*)(ws + 360 * MiB);
  __bf16* wto    = (__bf16*)(ws + 362 * MiB);
  __bf16* wt1    = (__bf16*)(ws + 364 * MiB);   // 8 MiB
  __bf16* wt2    = (__bf16*)(ws + 372 * MiB);   // 8 MiB  -> total 380 MiB
  __bf16* probs  = rkbf;
  float*  ffn1   = scores;

  // 0. one-time bf16 weight transposes (Wt[N,K])
  wprep_kernel<<<dim3(32, 32),  256, 0, stream>>>(Wq, wtq, 1024, 1024);
  wprep_kernel<<<dim3(32, 32),  256, 0, stream>>>(Wk, wtk, 1024, 1024);
  wprep_kernel<<<dim3(32, 32),  256, 0, stream>>>(Wv, wtv, 1024, 1024);
  wprep_kernel<<<dim3(32, 32),  256, 0, stream>>>(Wo, wto, 1024, 1024);
  wprep_kernel<<<dim3(32, 128), 256, 0, stream>>>(W1, wt1, 1024, 4096);
  wprep_kernel<<<dim3(128, 32), 256, 0, stream>>>(W2, wt2, 4096, 1024);
  // 1. pre-LN
  ln_kernel<<<2048, 256, 0, stream>>>(x, g1, b1, h);
  // 2. relation projections (dominant HBM stream)
  rproj_kernel<<<8192, 256, 0, stream>>>(r_k, Wr, br, rkbf);
  rproj_kernel<<<8192, 256, 0, stream>>>(r_v, Wr, br, rvbf);
  // 3. QKV projections
  gemm_kernel<false,false,true><<<dim3(8, 32), 256, 0, stream>>>(h, wtq, bq, nullptr, nullptr, qb, 2048, 1024, 1024);
  gemm_kernel<false,false,true><<<dim3(8, 32), 256, 0, stream>>>(h, wtk, bk, nullptr, nullptr, kb_, 2048, 1024, 1024);
  gemm_kernel<false,false,true><<<dim3(8, 32), 256, 0, stream>>>(h, wtv, bv, nullptr, nullptr, vb, 2048, 1024, 1024);
  // 4. scores = q.k^T, then += q.rk
  scores_qk_kernel<<<dim3(64, 32), 256, 0, stream>>>(qb, kb_, scores);
  scores_qr_kernel<<<2048, 256, 0, stream>>>(qb, rkbf, scores);
  // 5. softmax -> bf16 probs (aliases rkbf, now dead)
  softmax_kernel<<<4096, 256, 0, stream>>>(scores, mask, probs);
  // 6. ctx = a.v, then += a.rv
  ctx_av_kernel<<<dim3(64, 4), 256, 0, stream>>>(probs, vb, ctx);
  ctx_ar_kernel<<<256, 256, 0, stream>>>(probs, rvbf, ctx);
  // 7. output projection + residual(x)
  gemm_kernel<true,false,false><<<dim3(8, 32), 256, 0, stream>>>(ctx, wto, bo, x, outbuf, nullptr, 2048, 1024, 1024);
  // 8. LN2
  ln_kernel<<<2048, 256, 0, stream>>>(outbuf, g2, b2, h);
  // 9. FFN
  gemm_kernel<false,true,false><<<dim3(32, 32), 256, 0, stream>>>(h, wt1, c1, nullptr, ffn1, nullptr, 2048, 4096, 1024);
  gemm_kernel<true,false,false><<<dim3(8, 32), 256, 0, stream>>>(ffn1, wt2, c2, outbuf, (float*)d_out, nullptr, 2048, 1024, 4096);
}
